// PointTransformerBlock_10333691314739
// MI455X (gfx1250) — compile-verified, matrix operand order checked by reading, not run on hardware
//
#include <hip/hip_runtime.h>
#include <hip/hip_bf16.h>
#include <math.h>

// ---------------------------------------------------------------------------
// Point Transformer block, fused for MI455X (gfx1250, wave32, WMMA).
// ~39 GFLOP of per-pair MLP GEMMs -> bf16 v_wmma_f32_16x16x32_bf16.
// All pairwise intermediates LDS-resident; fragment feeds are b128 LDS ops:
//   A-fragments: 2x ds_load_b128 from row-major tiles (layout runs are
//                contiguous per ISA 7.12.2),
//   B-fragments: weights pre-swizzled once into fragment-major LDS banks.
// Flash-style channel-wise softmax, split over j with a combine kernel.
// ---------------------------------------------------------------------------

typedef __bf16 bf16_t;
typedef __attribute__((ext_vector_type(16))) __bf16 v16bf;
typedef __attribute__((ext_vector_type(8)))  __bf16 v8bf;
typedef __attribute__((ext_vector_type(8)))  float  v8f;

#define DIMC   64
#define POSH   64
#define ATTNH  256
#define BB     2
#define NN     512
#define ITILE  16
#define JTILE  16
#define NCHUNK 4
#define JCHUNK (NN / NCHUNK)      // 128 j per workgroup
#define THREADS 256
#define NWAVE  8

__device__ inline bf16_t f2bf(float f) { return (bf16_t)f; }

// --- A fragment (bf16 16x32) from row-major LDS tile: two b128 loads -------
// Lane l: row m = l%16. half = l/16. Elements [0..7] = row[k0..k0+7] with
// k0 = 8*half; elements [8..15] = row[16+8*half .. 23+8*half].
__device__ inline v16bf lds_load_A16x32(const bf16_t* __restrict__ src, int ld) {
  const int lane = threadIdx.x & 31;
  const int m    = lane & 15;
  const int half = (lane >> 4) & 1;
  const v8bf lo = *(const v8bf*)(src + m * ld + half * 8);
  const v8bf hi = *(const v8bf*)(src + m * ld + 16 + half * 8);
  return __builtin_shufflevector(lo, hi, 0, 1, 2, 3, 4, 5, 6, 7,
                                 8, 9, 10, 11, 12, 13, 14, 15);
}

// --- B fragment from a pre-swizzled fragment-major bank: two b128 loads ----
// Tile layout: frag[lane*16 + e], 512 bf16 (1KB) per 32x16 tile.
__device__ inline v16bf lds_load_Bfrag(const bf16_t* __restrict__ tile) {
  const int lane = threadIdx.x & 31;
  const bf16_t* p = tile + lane * 16;
  const v8bf lo = *(const v8bf*)p;
  const v8bf hi = *(const v8bf*)(p + 8);
  return __builtin_shufflevector(lo, hi, 0, 1, 2, 3, 4, 5, 6, 7,
                                 8, 9, 10, 11, 12, 13, 14, 15);
}

__device__ inline v8f wmma_bf16(v16bf a, v16bf b, v8f c) {
  return __builtin_amdgcn_wmma_f32_16x16x32_bf16(false, a, false, b, (short)0, c,
                                                 false, false);
}

// Swizzle a row-major f32 weight [K x N] into bf16 fragment-major B tiles.
// dst tile index t = kt*NT + nt; element (lane, e) = W[k][n] with
// k = kt*32 + ((e<8)? e : e+8) + 8*(lane/16), n = nt*16 + lane%16.
__device__ inline void fill_Bfrag(bf16_t* __restrict__ dst, const float* __restrict__ W,
                                  int N, int KT, int NT, int tid) {
  const int total = KT * NT * 512;
  for (int idx = tid; idx < total; idx += THREADS) {
    const int e    = idx & 15;
    const int lane = (idx >> 4) & 31;
    const int tile = idx >> 9;
    const int nt   = tile % NT;
    const int kt   = tile / NT;
    const int k    = kt * 32 + (((e < 8) ? e : e + 8) + (lane >> 4) * 8);
    const int n    = nt * 16 + (lane & 15);
    dst[idx] = f2bf(W[k * N + n]);
  }
}

// ---------------------------------------------------------------------------
// Kernel 1: x1 = x@prev_w+b ; q/k/v = x1@{q,k,v}_w   (tiny: 1024x64x64 GEMMs)
// ---------------------------------------------------------------------------
__global__ void pt_prep(const float* __restrict__ x,
                        const float* __restrict__ prev_w, const float* __restrict__ prev_b,
                        const float* __restrict__ q_w, const float* __restrict__ k_w,
                        const float* __restrict__ v_w,
                        float* __restrict__ x1, float* __restrict__ q,
                        float* __restrict__ k, float* __restrict__ v) {
  __shared__ float xs[DIMC];
  __shared__ float x1s[DIMC];
  const int row = blockIdx.x;            // b*N + n
  const int t   = threadIdx.x;           // 0..63
  xs[t] = x[row * DIMC + t];
  __syncthreads();
  float acc = prev_b[t];
#pragma unroll 16
  for (int c = 0; c < DIMC; ++c) acc += xs[c] * prev_w[c * DIMC + t];
  x1s[t] = acc;
  x1[row * DIMC + t] = acc;
  __syncthreads();
  float qa = 0.f, ka = 0.f, va = 0.f;
#pragma unroll 16
  for (int c = 0; c < DIMC; ++c) {
    const float xv = x1s[c];
    qa += xv * q_w[c * DIMC + t];
    ka += xv * k_w[c * DIMC + t];
    va += xv * v_w[c * DIMC + t];
  }
  q[row * DIMC + t] = qa;
  k[row * DIMC + t] = ka;
  v[row * DIMC + t] = va;
}

// ---------------------------------------------------------------------------
// Kernel 2: fused pairwise MLPs + online channel-wise softmax partials.
// ---------------------------------------------------------------------------
constexpr size_t SMEM_MAIN =
    size_t(2 * 16) * 512 * 2 +      // attn_w1 B-frag bank (KT=2, NT=16)
    size_t(8 * 4) * 512 * 2 +       // attn_w2 B-frag bank (KT=8, NT=4)
    size_t(2 * 4) * 512 * 2 +       // pos_w2  B-frag bank (KT=2, NT=4)
    3 * POSH * 4 +                  // pos_w1   f32
    POSH * 4 + DIMC * 4 +           // pos_b1, pos_b2
    ATTNH * 4 + DIMC * 4 +          // attn_b1, attn_b2
    ITILE * DIMC * 4 +              // q  rows (f32)
    JTILE * DIMC * 4 +              // k  rows (f32)
    JTILE * DIMC * 4 +              // v  rows (f32)
    ITILE * 16 + JTILE * 16 +       // pos_i, pos_j (padded stride 4)
    256 * POSH * 2 +                // hidden1 bf16 [256 pairs x 64]
    256 * DIMC * 2 +                // t bf16 [256 x 64]
    256 * DIMC * 4 +                // v2 f32 [256 x 64]
    size_t(NWAVE) * ITILE * 32 * 2; // per-wave C->A bounce scratch
// = 228352 bytes, fits the 320KB WGP LDS.

__global__ __launch_bounds__(THREADS, 1)
void pt_attn(const float* __restrict__ pos,
             const float* __restrict__ pos_w1, const float* __restrict__ pos_b1,
             const float* __restrict__ pos_w2, const float* __restrict__ pos_b2,
             const float* __restrict__ attn_w1, const float* __restrict__ attn_b1,
             const float* __restrict__ attn_w2, const float* __restrict__ attn_b2,
             const float* __restrict__ qg, const float* __restrict__ kg,
             const float* __restrict__ vg,
             float* __restrict__ ws_m, float* __restrict__ ws_l,
             float* __restrict__ ws_a) {
  extern __shared__ char smem_raw[];
  char* sp = smem_raw;
  auto carve = [&](size_t bytes) { char* r = sp; sp += (bytes + 15) & ~size_t(15); return r; };

  bf16_t* w1f  = (bf16_t*)carve(size_t(2 * 16) * 512 * 2);
  bf16_t* w2f  = (bf16_t*)carve(size_t(8 * 4) * 512 * 2);
  bf16_t* pw2f = (bf16_t*)carve(size_t(2 * 4) * 512 * 2);
  float*  pw1s = (float*) carve(3 * POSH * 4);
  float*  pb1s = (float*) carve(POSH * 4);
  float*  pb2s = (float*) carve(DIMC * 4);
  float*  ab1s = (float*) carve(ATTNH * 4);
  float*  ab2s = (float*) carve(DIMC * 4);
  float*  qis  = (float*) carve(ITILE * DIMC * 4);
  float*  kjs  = (float*) carve(JTILE * DIMC * 4);
  float*  vjs  = (float*) carve(JTILE * DIMC * 4);
  float*  pis  = (float*) carve(ITILE * 16);
  float*  pjs  = (float*) carve(JTILE * 16);
  bf16_t* h1s  = (bf16_t*)carve(256 * POSH * 2);
  bf16_t* ts   = (bf16_t*)carve(256 * DIMC * 2);
  float*  v2s  = (float*) carve(256 * DIMC * 4);
  bf16_t* scr  = (bf16_t*)carve(size_t(NWAVE) * ITILE * 32 * 2);

  const int tid  = threadIdx.x;
  const int wave = tid >> 5;
  const int lane = tid & 31;
  const int half = (lane >> 4) & 1;
  const int ln   = lane & 15;

  const int bid   = blockIdx.x;
  const int b     = bid / ((NN / ITILE) * NCHUNK);
  const int rem   = bid % ((NN / ITILE) * NCHUNK);
  const int itile = rem / NCHUNK;
  const int chunk = rem % NCHUNK;
  const int i0    = itile * ITILE;

  // ---- stage weights (pre-swizzled to B-fragment banks) + biases + q tile --
  fill_Bfrag(w1f,  attn_w1, ATTNH, 2, 16, tid);
  fill_Bfrag(w2f,  attn_w2, DIMC,  8, 4,  tid);
  fill_Bfrag(pw2f, pos_w2,  DIMC,  2, 4,  tid);
  for (int idx = tid; idx < 3 * POSH; idx += THREADS) pw1s[idx] = pos_w1[idx];
  for (int idx = tid; idx < POSH;  idx += THREADS) pb1s[idx] = pos_b1[idx];
  for (int idx = tid; idx < ATTNH; idx += THREADS) ab1s[idx] = attn_b1[idx];
  for (int idx = tid; idx < DIMC;  idx += THREADS) { pb2s[idx] = pos_b2[idx]; ab2s[idx] = attn_b2[idx]; }
  ((float4*)qis)[tid] = ((const float4*)(qg + (b * NN + i0) * DIMC))[tid];
  for (int idx = tid; idx < ITILE * 3; idx += THREADS)
    pis[(idx / 3) * 4 + (idx % 3)] = pos[(b * NN + i0) * 3 + idx];
  __syncthreads();

  // ---- online softmax state: per wave, per M-tile {2w,2w+1}, per N-tile ----
  float m_run[2][4], l_run[2][4], a_run[2][4];
#pragma unroll
  for (int mi = 0; mi < 2; ++mi)
#pragma unroll
    for (int n = 0; n < 4; ++n) { m_run[mi][n] = -INFINITY; l_run[mi][n] = 0.f; a_run[mi][n] = 0.f; }

  for (int jt = 0; jt < JCHUNK / JTILE; ++jt) {
    const int j0 = chunk * JCHUNK + jt * JTILE;

    // ---- stage A: k/v/pos_j tiles into LDS (b128 global -> b128 LDS) ----
    ((float4*)kjs)[tid] = ((const float4*)(kg + (b * NN + j0) * DIMC))[tid];
    ((float4*)vjs)[tid] = ((const float4*)(vg + (b * NN + j0) * DIMC))[tid];
    for (int idx = tid; idx < JTILE * 3; idx += THREADS)
      pjs[(idx / 3) * 4 + (idx % 3)] = pos[(b * NN + j0) * 3 + idx];
    if (jt + 1 < JCHUNK / JTILE && tid < JTILE) {   // prefetch next tile's rows
      const int jn = chunk * JCHUNK + (jt + 1) * JTILE;
      __builtin_prefetch(&kg[(b * NN + jn + tid) * DIMC], 0, 1);
      __builtin_prefetch(&vg[(b * NN + jn + tid) * DIMC], 0, 1);
    }
    __syncthreads();

    // ---- pos-MLP layer 1 (K=3: VALU), thread tid owns pair p = tid ----
    {
      const int i_l = tid >> 4, j_l = tid & 15;
      const float r0 = pis[i_l * 4 + 0] - pjs[j_l * 4 + 0];
      const float r1 = pis[i_l * 4 + 1] - pjs[j_l * 4 + 1];
      const float r2 = pis[i_l * 4 + 2] - pjs[j_l * 4 + 2];
#pragma unroll
      for (int h0 = 0; h0 < POSH; h0 += 8) {
        v8bf pack;
#pragma unroll
        for (int u = 0; u < 8; ++u) {
          const int h = h0 + u;
          const float hv = pb1s[h] + r0 * pw1s[h] + r1 * pw1s[POSH + h] + r2 * pw1s[2 * POSH + h];
          pack[u] = f2bf(fmaxf(hv, 0.f));
        }
        *(v8bf*)(h1s + tid * POSH + h0) = pack;   // ds_store_b128
      }
    }
    asm volatile("s_wait_dscnt 0" ::: "memory");   // h1s is intra-wave (rows 32w..32w+31)

    // ---- rpe = h1 @ pos_w2 (WMMA) fused into t = q-k+rpe, v2 = v+rpe ----
#pragma unroll
    for (int mi = 0; mi < 2; ++mi) {
      const int mt = wave * 2 + mi;
      const bf16_t* h1p = h1s + mt * ITILE * POSH;
      const v16bf a0 = lds_load_A16x32(h1p, POSH);
      const v16bf a1 = lds_load_A16x32(h1p + 32, POSH);
#pragma unroll
      for (int n = 0; n < 4; ++n) {
        v8f acc = {};
        acc = wmma_bf16(a0, lds_load_Bfrag(pw2f + (0 * 4 + n) * 512), acc);
        acc = wmma_bf16(a1, lds_load_Bfrag(pw2f + (1 * 4 + n) * 512), acc);
        const int   d  = n * 16 + ln;
        const float qv = qis[mt * DIMC + d];
        const float b2 = pb2s[d];
#pragma unroll
        for (int r = 0; r < 8; ++r) {
          const int   m    = r + half * 8;          // j_local
          const float rpe  = acc[r] + b2;
          const int   pidx = mt * 16 + m;
          ts[pidx * DIMC + d]  = f2bf(qv - kjs[m * DIMC + d] + rpe);
          v2s[pidx * DIMC + d] = vjs[m * DIMC + d] + rpe;
        }
      }
    }
    asm volatile("s_wait_dscnt 0" ::: "memory");

    // ---- sim = relu(t @ attn_w1 + b1) @ attn_w2 + b2, then online softmax ----
#pragma unroll
    for (int mi = 0; mi < 2; ++mi) {
      const int mt = wave * 2 + mi;
      const bf16_t* tp = ts + mt * ITILE * DIMC;
      const v16bf ta0 = lds_load_A16x32(tp, DIMC);
      const v16bf ta1 = lds_load_A16x32(tp + 32, DIMC);
      bf16_t* myscr = scr + wave * ITILE * 32;
      v8f sacc[4];
      {
        v8f z = {};
        sacc[0] = z; sacc[1] = z; sacc[2] = z; sacc[3] = z;
      }
      for (int kc = 0; kc < 8; ++kc) {              // hidden dim in chunks of 32
#pragma unroll
        for (int xh = 0; xh < 2; ++xh) {
          const int hn = kc * 2 + xh;               // hidden N-tile (0..15)
          v8f hacc = {};
          hacc = wmma_bf16(ta0, lds_load_Bfrag(w1f + (0 * 16 + hn) * 512), hacc);
          hacc = wmma_bf16(ta1, lds_load_Bfrag(w1f + (1 * 16 + hn) * 512), hacc);
          const float hb = ab1s[hn * 16 + ln];
#pragma unroll
          for (int r = 0; r < 8; ++r) {
            const int m = r + half * 8;
            myscr[m * 32 + xh * 16 + ln] = f2bf(fmaxf(hacc[r] + hb, 0.f));
          }
        }
        asm volatile("s_wait_dscnt 0" ::: "memory");  // C->A bounce via LDS
        const v16bf ha = lds_load_A16x32(myscr, 32);
#pragma unroll
        for (int n = 0; n < 4; ++n)
          sacc[n] = wmma_bf16(ha, lds_load_Bfrag(w2f + (kc * 4 + n) * 512), sacc[n]);
      }
      // flash-style update over this j-tile, per channel d (lane pairs l/l+16
      // share (i,d) and keep identical state via shfl_xor reductions).
#pragma unroll
      for (int n = 0; n < 4; ++n) {
        const int   d  = n * 16 + ln;
        const float b2 = ab2s[d];
        float s[8];
        float tmax = -INFINITY;
#pragma unroll
        for (int r = 0; r < 8; ++r) { s[r] = sacc[n][r] + b2; tmax = fmaxf(tmax, s[r]); }
        tmax = fmaxf(tmax, __shfl_xor(tmax, 16, 32));
        const float newm  = fmaxf(m_run[mi][n], tmax);
        const float scale = __expf(m_run[mi][n] - newm);
        float psum = 0.f, pacc = 0.f;
#pragma unroll
        for (int r = 0; r < 8; ++r) {
          const int   m = r + half * 8;
          const float e = __expf(s[r] - newm);
          psum += e;
          pacc += e * v2s[(mt * 16 + m) * DIMC + d];
        }
        psum += __shfl_xor(psum, 16, 32);
        pacc += __shfl_xor(pacc, 16, 32);
        l_run[mi][n] = l_run[mi][n] * scale + psum;
        a_run[mi][n] = a_run[mi][n] * scale + pacc;
        m_run[mi][n] = newm;
      }
    }
    __syncthreads();  // before next iteration overwrites kjs/vjs/pjs/h1s
  }

  // ---- write split-j softmax partials ----
  if (half == 0) {
#pragma unroll
    for (int mi = 0; mi < 2; ++mi) {
      const int ig = i0 + wave * 2 + mi;
#pragma unroll
      for (int n = 0; n < 4; ++n) {
        const int d   = n * 16 + ln;
        const int idx = ((b * NN + ig) * DIMC + d) * NCHUNK + chunk;
        ws_m[idx] = m_run[mi][n];
        ws_l[idx] = l_run[mi][n];
        ws_a[idx] = a_run[mi][n];
      }
    }
  }
}

// ---------------------------------------------------------------------------
// Kernel 3: combine j-chunk partials; out = (acc/l) @ fin_w + fin_b + x1
// ---------------------------------------------------------------------------
__global__ void pt_finalize(const float* __restrict__ ws_m, const float* __restrict__ ws_l,
                            const float* __restrict__ ws_a, const float* __restrict__ x1,
                            const float* __restrict__ fin_w, const float* __restrict__ fin_b,
                            float* __restrict__ out) {
  __shared__ float aggs[DIMC];
  const int row  = blockIdx.x;            // b*N + i
  const int t    = threadIdx.x;           // 0..63 = channel d
  const int base = (row * DIMC + t) * NCHUNK;
  float M = -INFINITY;
#pragma unroll
  for (int c = 0; c < NCHUNK; ++c) M = fmaxf(M, ws_m[base + c]);
  float L = 0.f, A = 0.f;
#pragma unroll
  for (int c = 0; c < NCHUNK; ++c) {
    const float sc = __expf(ws_m[base + c] - M);
    L += ws_l[base + c] * sc;
    A += ws_a[base + c] * sc;
  }
  aggs[t] = A / L;
  __syncthreads();
  float o = fin_b[t];
#pragma unroll 16
  for (int c = 0; c < DIMC; ++c) o += aggs[c] * fin_w[c * DIMC + t];
  out[row * DIMC + t] = o + x1[row * DIMC + t];
}

// ---------------------------------------------------------------------------
extern "C" void kernel_launch(void* const* d_in, const int* in_sizes, int n_in,
                              void* d_out, int out_size, void* d_ws, size_t ws_size,
                              hipStream_t stream) {
  (void)in_sizes; (void)n_in; (void)out_size; (void)ws_size;
  const float* x       = (const float*)d_in[0];
  const float* pos     = (const float*)d_in[1];
  const float* prev_w  = (const float*)d_in[2];
  const float* prev_b  = (const float*)d_in[3];
  const float* q_w     = (const float*)d_in[4];
  const float* k_w     = (const float*)d_in[5];
  const float* v_w     = (const float*)d_in[6];
  const float* pos_w1  = (const float*)d_in[7];
  const float* pos_b1  = (const float*)d_in[8];
  const float* pos_w2  = (const float*)d_in[9];
  const float* pos_b2  = (const float*)d_in[10];
  const float* attn_w1 = (const float*)d_in[11];
  const float* attn_b1 = (const float*)d_in[12];
  const float* attn_w2 = (const float*)d_in[13];
  const float* attn_b2 = (const float*)d_in[14];
  const float* fin_w   = (const float*)d_in[15];
  const float* fin_b   = (const float*)d_in[16];
  float* out = (float*)d_out;
  float* ws  = (float*)d_ws;

  const size_t ROWS = size_t(BB) * NN;          // 1024
  float* ws_x1 = ws;
  float* ws_q  = ws + ROWS * DIMC;
  float* ws_k  = ws + 2 * ROWS * DIMC;
  float* ws_v  = ws + 3 * ROWS * DIMC;
  float* ws_m  = ws + 4 * ROWS * DIMC;          // [B,N,D,NCHUNK] planes
  float* ws_l  = ws_m + ROWS * DIMC * NCHUNK;
  float* ws_a  = ws_l + ROWS * DIMC * NCHUNK;

  pt_prep<<<dim3(BB * NN), dim3(DIMC), 0, stream>>>(
      x, prev_w, prev_b, q_w, k_w, v_w, ws_x1, ws_q, ws_k, ws_v);

  pt_attn<<<dim3(BB * (NN / ITILE) * NCHUNK), dim3(THREADS), SMEM_MAIN, stream>>>(
      pos, pos_w1, pos_b1, pos_w2, pos_b2, attn_w1, attn_b1, attn_w2, attn_b2,
      ws_q, ws_k, ws_v, ws_m, ws_l, ws_a);

  pt_finalize<<<dim3(BB * NN), dim3(DIMC), 0, stream>>>(
      ws_m, ws_l, ws_a, ws_x1, fin_w, fin_b, out);

  // output tuple: (x_out [B,N,D], pos [B,N,3]) concatenated flat
  hipMemcpyAsync(out + ROWS * DIMC, pos, ROWS * 3 * sizeof(float),
                 hipMemcpyDeviceToDevice, stream);
}
